// Vrnn_74208444940736
// MI455X (gfx1250) — compile-verified, matrix operand order checked by reading, not run on hardware
//
#include <hip/hip_runtime.h>
#include <hip/hip_bf16.h>

typedef _Float16 f16;
typedef __attribute__((ext_vector_type(16))) _Float16 v16h;
typedef __attribute__((ext_vector_type(8)))  float    v8f;

#define B_ 64
#define T_ 128

__device__ __forceinline__ float softplus_f(float x) {
    return fmaxf(x, 0.0f) + log1pf(expf(-fabsf(x)));   // stable log(1+exp(x))
}
__device__ __forceinline__ float sigmoid_f(float x) {
    return 1.0f / (1.0f + expf(-x));
}

// 16-byte async copy global -> LDS (CDNA5 GLOBAL_LOAD_ASYNC_TO_LDS_B128, ASYNCcnt)
__device__ __forceinline__ void async_cp16(unsigned lds_off, const f16* g) {
    unsigned long long ga = (unsigned long long)(uintptr_t)g;
    asm volatile("global_load_async_to_lds_b128 %0, %1, off"
                 :: "v"(lds_off), "v"(ga) : "memory");
}

// GEMM problem descriptor:  out = act( [A0 | A1] @ Wt^T + bias )
//   A0: M x K0 (row-major, lda0), A1: M x K1 (row-major, lda1) -- split-K concat
//   Wt: N x K row-major (pre-transposed weight), K = K0 + K1
//   out16 (f16, ldc16) and/or out32 (f32, ldc32), either may be null
//   act: 0 = none, 1 = relu, 2 = softplus
// Requires M % 64 == 0, N % 64 == 0, K0/K1 % 32 == 0.
struct GemmDesc {
    const f16*   A0;  int lda0; int K0;
    const f16*   A1;  int lda1; int K1;
    const f16*   Wt;
    const float* bias;
    f16*         out16; int ldc16;
    float*       out32; int ldc32;
    int nblocks;        // N / 64
    int act;
};

// ---------------------------------------------------------------------------
// Dual-problem WMMA f16 GEMM.  Block = 128 threads (4 waves) owns a 64x64
// output tile of either problem `da` (blockIdx < blocksA) or `db`.  The K
// loop runs a double-buffered async-to-LDS pipeline: panel k+1 streams in
// (ASYNCcnt) while the 4 waves consume panel k from LDS via ds_load_b128
// into v_wmma_f32_16x16x32_f16.
// ---------------------------------------------------------------------------
__global__ __launch_bounds__(128) void k_gemm2(GemmDesc da, GemmDesc db, int blocksA)
{
    __shared__ __align__(16) f16 sA[2][64 * 32];   // [buf][m-row][k] 2 x 4KB
    __shared__ __align__(16) f16 sB[2][64 * 32];   // [buf][n-row][k] 2 x 4KB

    GemmDesc d; int bid;
    if ((int)blockIdx.x < blocksA) { d = da; bid = blockIdx.x; }
    else                           { d = db; bid = blockIdx.x - blocksA; }

    const int tid  = threadIdx.x;
    const int wave = tid >> 5;
    const int lane = tid & 31;
    const int m0   = (bid / d.nblocks) << 6;    // block M base
    const int n0   = (bid % d.nblocks) << 6;    // block N base
    const int ll   = lane & 15;                 // row (A) / col (B,C) within tile
    const int hi   = lane >> 4;                 // K-half selector per ISA layout
    const int K    = d.K0 + d.K1;
    const int nk   = K >> 5;

    // Staging: 256 chunks of 16B per panel; this thread moves chunks tid, tid+128.
    // chunk c: row = c>>2, kq = c&3 ; LDS byte off = c*16 ; elem off = row*ld + kq*8
    const int r0 = tid >> 2,         q0 = tid & 3;
    const int r1 = (tid + 128) >> 2, q1 = (tid + 128) & 3;

    const unsigned sAo = (unsigned)(uintptr_t)&sA[0][0];  // low 32 bits = LDS offset
    const unsigned sBo = (unsigned)(uintptr_t)&sB[0][0];

    // issue one 8KB panel pair (A 64x32, B 64x32) into buffer `buf`
    auto issue = [&](int k, int buf) {
        const f16* Ap; int la; int kk;
        if (k < d.K0) { Ap = d.A0; la = d.lda0; kk = k; }
        else          { Ap = d.A1; la = d.lda1; kk = k - d.K0; }
        const unsigned bo = (unsigned)buf * 4096u;
        async_cp16(sAo + bo + (unsigned)tid * 16,
                   Ap + (size_t)(m0 + r0) * la + kk + q0 * 8);
        async_cp16(sAo + bo + (unsigned)(tid + 128) * 16,
                   Ap + (size_t)(m0 + r1) * la + kk + q1 * 8);
        async_cp16(sBo + bo + (unsigned)tid * 16,
                   d.Wt + (size_t)(n0 + r0) * K + k + q0 * 8);
        async_cp16(sBo + bo + (unsigned)(tid + 128) * 16,
                   d.Wt + (size_t)(n0 + r1) * K + k + q1 * 8);
    };

    v8f zero = {0.f, 0.f, 0.f, 0.f, 0.f, 0.f, 0.f, 0.f};
    v8f acc[4] = {zero, zero, zero, zero};

    issue(0, 0);                                    // prefetch panel 0
    for (int i = 0; i < nk; ++i) {
        __syncthreads();   // all waves done reading buf[(i+1)&1] (iteration i-1)
        if (i + 1 < nk) {
            issue((i + 1) << 5, (i + 1) & 1);       // prefetch next panel
            // panels complete in order: cnt<=4 => panel i's 4 ops are done
            asm volatile("s_wait_asynccnt 0x4" ::: "memory");
        } else {
            asm volatile("s_wait_asynccnt 0x0" ::: "memory");
        }
        __syncthreads();   // panel i visible to all waves

        const f16* bufA = &sA[i & 1][0];
        const f16* bufB = &sB[i & 1][0];

        // A fragment 16x32: lane ll = row of wave's 16-row slice, hi = K halves
        union { v16h v; uint4 q[2]; } af;
        const f16* arow = bufA + (wave * 16 + ll) * 32;
        af.q[0] = *(const uint4*)(arow + hi * 8);
        af.q[1] = *(const uint4*)(arow + 16 + hi * 8);

        #pragma unroll
        for (int j = 0; j < 4; ++j) {
            union { v16h v; uint4 q[2]; } bf;
            const f16* brow = bufB + (j * 16 + ll) * 32;
            bf.q[0] = *(const uint4*)(brow + hi * 8);
            bf.q[1] = *(const uint4*)(brow + 16 + hi * 8);
            acc[j] = __builtin_amdgcn_wmma_f32_16x16x32_f16(
                false, af.v, false, bf.v, (short)0, acc[j], false, false);
        }
    }

    // C/D layout: VGPR r -> row base + r + hi*8, lane ll -> column
    const int mrow = m0 + wave * 16;
    #pragma unroll
    for (int j = 0; j < 4; ++j) {
        const int ncol = n0 + j * 16 + ll;
        const float bvl = d.bias ? d.bias[ncol] : 0.0f;
        #pragma unroll
        for (int r = 0; r < 8; ++r) {
            const int row = mrow + r + hi * 8;
            float v = acc[j][r] + bvl;
            if (d.act == 1)      v = fmaxf(v, 0.0f);
            else if (d.act == 2) v = softplus_f(v);
            if (d.out16) d.out16[(size_t)row * d.ldc16 + ncol] = (f16)v;
            if (d.out32) d.out32[(size_t)row * d.ldc32 + ncol] = v;
        }
    }
}

// fp32 -> f16 cast (same layout)
__global__ void k_cvt(const float* __restrict__ x, f16* __restrict__ y, int n) {
    int i = blockIdx.x * blockDim.x + threadIdx.x;
    if (i < n) y[i] = (f16)x[i];
}

// fp32 W (K x N) -> f16 Wt (N x K)  (transpose + cast)
__global__ void k_cvt_t(const float* __restrict__ W, f16* __restrict__ Wt, int K, int N) {
    int i = blockIdx.x * blockDim.x + threadIdx.x;
    if (i < K * N) {
        int n = i / K, k = i % K;
        Wt[(size_t)n * K + k] = (f16)W[(size_t)k * N + n];
    }
}

// z_t = mu + softplus(slin) * eps   (writes fp32 z to d_out region + f16 copy)
__global__ void k_sample_z(const float* __restrict__ mu, const float* __restrict__ slin,
                           const float* __restrict__ eps_t, int eps_stride,
                           float* __restrict__ zout, int z_stride,
                           f16* __restrict__ z16)
{
    int i = blockIdx.x * blockDim.x + threadIdx.x;   // B_ * 128
    if (i >= B_ * 128) return;
    int b = i >> 7, c = i & 127;
    float sig = softplus_f(slin[i]);
    float z = mu[i] + sig * eps_t[(size_t)b * eps_stride + c];
    zout[(size_t)b * z_stride + c] = z;
    z16[i] = (f16)z;
}

// keras GRUCell combine (z/r/n order): h' = z*h + (1-z)*tanh(xn + r*hn)
__global__ void k_gru(const float* __restrict__ gx, const float* __restrict__ gh,
                      const float* __restrict__ hst,
                      float* __restrict__ hnext, f16* __restrict__ hnext16,
                      f16* __restrict__ hall16, int hall_stride)
{
    int i = blockIdx.x * blockDim.x + threadIdx.x;   // B_ * 1024
    if (i >= B_ * 1024) return;
    int b = i >> 10, c = i & 1023;
    const float* gxr = gx + (size_t)b * 3072;
    const float* ghr = gh + (size_t)b * 3072;
    float zg = sigmoid_f(gxr[c] + ghr[c]);
    float rg = sigmoid_f(gxr[1024 + c] + ghr[1024 + c]);
    float ng = tanhf(gxr[2048 + c] + rg * ghr[2048 + c]);
    float h  = hst[i];
    float o  = zg * h + (1.0f - zg) * ng;
    hnext[i]   = o;
    hnext16[i] = (f16)o;
    if (hall16) hall16[(size_t)b * hall_stride + c] = (f16)o;
}

// ---------------------------------------------------------------------------

static inline size_t align256(size_t x) { return (x + 255) & ~(size_t)255; }

extern "C" void kernel_launch(void* const* d_in, const int* in_sizes, int n_in,
                              void* d_out, int out_size, void* d_ws, size_t ws_size,
                              hipStream_t stream)
{
    (void)in_sizes; (void)n_in; (void)out_size; (void)ws_size;

    const float* inp  = (const float*)d_in[0];
    const float* eps  = (const float*)d_in[1];
    const float* xtW1 = (const float*)d_in[2];  const float* xtb1 = (const float*)d_in[3];
    const float* xtW2 = (const float*)d_in[4];  const float* xtb2 = (const float*)d_in[5];
    const float* ztW1 = (const float*)d_in[6];  const float* ztb1 = (const float*)d_in[7];
    const float* ztW2 = (const float*)d_in[8];  const float* ztb2 = (const float*)d_in[9];
    const float* phW1 = (const float*)d_in[10]; const float* phb1 = (const float*)d_in[11];
    const float* phW2 = (const float*)d_in[12]; const float* phb2 = (const float*)d_in[13];
    const float* phmW = (const float*)d_in[14]; const float* phmb = (const float*)d_in[15];
    const float* phsW = (const float*)d_in[16]; const float* phsb = (const float*)d_in[17];
    // indices 18..25: prior network -- not part of the returned outputs, skipped
    const float* thW1 = (const float*)d_in[26]; const float* thb1 = (const float*)d_in[27];
    const float* thW2 = (const float*)d_in[28]; const float* thb2 = (const float*)d_in[29];
    const float* thmW = (const float*)d_in[30]; const float* thmb = (const float*)d_in[31];
    const float* thsW = (const float*)d_in[32]; const float* thsb = (const float*)d_in[33];
    const float* g1Wi = (const float*)d_in[34]; const float* g1Wh = (const float*)d_in[35];
    const float* g1bi = (const float*)d_in[36]; const float* g1bh = (const float*)d_in[37];
    const float* g2Wi = (const float*)d_in[38]; const float* g2Wh = (const float*)d_in[39];
    const float* g2bi = (const float*)d_in[40]; const float* g2bh = (const float*)d_in[41];
    const float* g3Wi = (const float*)d_in[42]; const float* g3Wh = (const float*)d_in[43];
    const float* g3bi = (const float*)d_in[44]; const float* g3bh = (const float*)d_in[45];

    const int BT = B_ * T_;                         // 8192

    // ---- workspace bump allocator -------------------------------------
    char* p = (char*)d_ws;
    auto alloc = [&](size_t bytes) -> char* { char* r = p; p += align256(bytes); return r; };

    f16* in16 = (f16*)alloc((size_t)BT * 128 * 2);
    f16* X16  = (f16*)alloc((size_t)BT * 512 * 2);    // x2s features (B,T,512)
    f16* Zp16 = (f16*)alloc((size_t)BT * 512 * 2);    // z2s features (B,T,512); also X1 temp
    f16* H16  = (f16*)alloc((size_t)BT * 1024 * 2);   // states (B,T,1024)

    f16* w_xt1 = (f16*)alloc((size_t)128  * 512  * 2);
    f16* w_xt2 = (f16*)alloc((size_t)512  * 512  * 2);
    f16* w_zt1 = (f16*)alloc((size_t)128  * 512  * 2);
    f16* w_zt2 = (f16*)alloc((size_t)512  * 512  * 2);
    f16* w_ph1 = (f16*)alloc((size_t)1536 * 1024 * 2);
    f16* w_ph2 = (f16*)alloc((size_t)1024 * 1024 * 2);
    f16* w_phm = (f16*)alloc((size_t)1024 * 128  * 2);
    f16* w_phs = (f16*)alloc((size_t)1024 * 128  * 2);
    f16* w_th1 = (f16*)alloc((size_t)1536 * 1024 * 2);
    f16* w_th2 = (f16*)alloc((size_t)1024 * 1024 * 2);
    f16* w_thm = (f16*)alloc((size_t)1024 * 128  * 2);
    f16* w_ths = (f16*)alloc((size_t)1024 * 128  * 2);
    f16* w_g1i = (f16*)alloc((size_t)1024 * 3072 * 2);
    f16* w_g1h = (f16*)alloc((size_t)1024 * 3072 * 2);
    f16* w_g2i = (f16*)alloc((size_t)1024 * 3072 * 2);
    f16* w_g2h = (f16*)alloc((size_t)1024 * 3072 * 2);
    f16* w_g3i = (f16*)alloc((size_t)1024 * 3072 * 2);
    f16* w_g3h = (f16*)alloc((size_t)1024 * 3072 * 2);

    // per-step scratch
    f16*   phi1_16 = (f16*)alloc((size_t)B_ * 1024 * 2);
    f16*   phi16   = (f16*)alloc((size_t)B_ * 1024 * 2);
    float* mu32    = (float*)alloc((size_t)B_ * 128 * 4);
    float* sl32    = (float*)alloc((size_t)B_ * 128 * 4);
    f16*   z16     = (f16*)alloc((size_t)B_ * 128 * 2);
    f16*   zp1_16  = (f16*)alloc((size_t)B_ * 512 * 2);
    float* gx32    = (float*)alloc((size_t)B_ * 3072 * 4);
    float* gh32    = (float*)alloc((size_t)B_ * 3072 * 4);
    float* h32     = (float*)alloc((size_t)B_ * 1024 * 4);
    f16*   h16     = (f16*)alloc((size_t)B_ * 1024 * 2);
    float* h1_32   = (float*)alloc((size_t)B_ * 1024 * 4);
    f16*   h1_16   = (f16*)alloc((size_t)B_ * 1024 * 2);
    float* h2_32   = (float*)alloc((size_t)B_ * 1024 * 4);
    f16*   h2_16   = (f16*)alloc((size_t)B_ * 1024 * 2);

    // decoder chunk buffers (M processed in 4 chunks of 2048 rows)
    const int CM = 2048;
    f16* t1_16 = (f16*)alloc((size_t)CM * 1024 * 2);
    f16* t2_16 = (f16*)alloc((size_t)CM * 1024 * 2);

    // ---- launch helpers ------------------------------------------------
    auto mkdesc = [](const f16* A0, int lda0, int K0, const f16* A1, int lda1, int K1,
                     const f16* Wt, const float* bias,
                     f16* o16, int ldc16, float* o32, int ldc32, int N, int act) {
        GemmDesc d;
        d.A0 = A0; d.lda0 = lda0; d.K0 = K0;
        d.A1 = A1; d.lda1 = lda1; d.K1 = K1;
        d.Wt = Wt; d.bias = bias;
        d.out16 = o16; d.ldc16 = ldc16;
        d.out32 = o32; d.ldc32 = ldc32;
        d.nblocks = N / 64; d.act = act;
        return d;
    };
    auto gemm1 = [&](const GemmDesc& d, int M) {
        int blocks = (M / 64) * d.nblocks;
        k_gemm2<<<blocks, 128, 0, stream>>>(d, d, blocks);
    };
    auto gemm2 = [&](const GemmDesc& d0, int M0, const GemmDesc& d1, int M1) {
        int b0 = (M0 / 64) * d0.nblocks;
        int b1 = (M1 / 64) * d1.nblocks;
        k_gemm2<<<b0 + b1, 128, 0, stream>>>(d0, d1, b0);
    };
    auto cvt_t = [&](const float* W, f16* Wt, int K, int N) {
        int n = K * N;
        k_cvt_t<<<(n + 255) / 256, 256, 0, stream>>>(W, Wt, K, N);
    };

    // ---- prologue: h0 = 0, weight conversion, input cast ---------------
    hipMemsetAsync(h32, 0, (size_t)B_ * 1024 * 4, stream);
    hipMemsetAsync(h16, 0, (size_t)B_ * 1024 * 2, stream);

    cvt_t(xtW1, w_xt1, 128, 512);   cvt_t(xtW2, w_xt2, 512, 512);
    cvt_t(ztW1, w_zt1, 128, 512);   cvt_t(ztW2, w_zt2, 512, 512);
    cvt_t(phW1, w_ph1, 1536, 1024); cvt_t(phW2, w_ph2, 1024, 1024);
    cvt_t(phmW, w_phm, 1024, 128);  cvt_t(phsW, w_phs, 1024, 128);
    cvt_t(thW1, w_th1, 1536, 1024); cvt_t(thW2, w_th2, 1024, 1024);
    cvt_t(thmW, w_thm, 1024, 128);  cvt_t(thsW, w_ths, 1024, 128);
    cvt_t(g1Wi, w_g1i, 1024, 3072); cvt_t(g1Wh, w_g1h, 1024, 3072);
    cvt_t(g2Wi, w_g2i, 1024, 3072); cvt_t(g2Wh, w_g2h, 1024, 3072);
    cvt_t(g3Wi, w_g3i, 1024, 3072); cvt_t(g3Wh, w_g3h, 1024, 3072);

    {
        int n = BT * 128;
        k_cvt<<<(n + 255) / 256, 256, 0, stream>>>(inp, in16, n);
    }

    // ---- X transform (time-parallel): X = relu(relu(in@xtW1+b)@xtW2+b) -
    // Zp16 doubles as the X1 temp (fully overwritten during the scan).
    gemm1(mkdesc(in16, 128, 128, nullptr, 0, 0, w_xt1, xtb1, Zp16, 512, nullptr, 0, 512, 1), BT);
    gemm1(mkdesc(Zp16, 512, 512, nullptr, 0, 0, w_xt2, xtb2, X16, 512, nullptr, 0, 512, 1), BT);

    // ---- sequential scan over T ----------------------------------------
    float* out_mu  = (float*)d_out;                 // (B,T,128)
    float* out_sig = out_mu + (size_t)BT * 128;     // (B,T,128)
    float* out_z   = out_sig + (size_t)BT * 128;    // (B,T,128)
    const int ldX = T_ * 512;                       // row stride of (B,T,512) slice at fixed t

    for (int t = 0; t < T_; ++t) {
        const f16* Xt  = X16 + (size_t)t * 512;
        f16*       Zpt = Zp16 + (size_t)t * 512;

        // phi = relu(relu([x_t|h]@phW1+b)@phW2+b)
        gemm1(mkdesc(Xt, ldX, 512, h16, 1024, 1024, w_ph1, phb1,
                     phi1_16, 1024, nullptr, 0, 1024, 1), B_);
        gemm1(mkdesc(phi1_16, 1024, 1024, nullptr, 0, 0, w_ph2, phb2,
                     phi16, 1024, nullptr, 0, 1024, 1), B_);
        // mu / sigma heads in one launch (linear, fp32 out)
        gemm2(mkdesc(phi16, 1024, 1024, nullptr, 0, 0, w_phm, phmb,
                     nullptr, 0, mu32, 128, 128, 0), B_,
              mkdesc(phi16, 1024, 1024, nullptr, 0, 0, w_phs, phsb,
                     nullptr, 0, sl32, 128, 128, 0), B_);
        // z_t = mu + softplus(slin)*eps   (also emit z output)
        k_sample_z<<<(B_ * 128 + 255) / 256, 256, 0, stream>>>(
            mu32, sl32, eps + (size_t)t * 128, T_ * 128,
            out_z + (size_t)t * 128, T_ * 128, z16);
        // zp = relu(relu(z@ztW1+b)@ztW2+b)  -> stored into Zp_all[:,t,:]
        gemm1(mkdesc(z16, 128, 128, nullptr, 0, 0, w_zt1, ztb1,
                     zp1_16, 512, nullptr, 0, 512, 1), B_);
        gemm1(mkdesc(zp1_16, 512, 512, nullptr, 0, 0, w_zt2, ztb2,
                     Zpt, ldX, nullptr, 0, 512, 1), B_);
        // GRU1: gx = [x_t|zp]@Wi+bi , gh = h@Wh+bh   (one launch)
        gemm2(mkdesc(Xt, ldX, 512, Zpt, ldX, 512, w_g1i, g1bi,
                     nullptr, 0, gx32, 3072, 3072, 0), B_,
              mkdesc(h16, 1024, 1024, nullptr, 0, 0, w_g1h, g1bh,
                     nullptr, 0, gh32, 3072, 3072, 0), B_);
        k_gru<<<(B_ * 1024 + 255) / 256, 256, 0, stream>>>(gx32, gh32, h32, h1_32, h1_16, nullptr, 0);
        // GRU2: input = state = h1
        gemm2(mkdesc(h1_16, 1024, 1024, nullptr, 0, 0, w_g2i, g2bi,
                     nullptr, 0, gx32, 3072, 3072, 0), B_,
              mkdesc(h1_16, 1024, 1024, nullptr, 0, 0, w_g2h, g2bh,
                     nullptr, 0, gh32, 3072, 3072, 0), B_);
        k_gru<<<(B_ * 1024 + 255) / 256, 256, 0, stream>>>(gx32, gh32, h1_32, h2_32, h2_16, nullptr, 0);
        // GRU3: input = state = h2 ; result is next h, archived in H_all[:,t,:]
        gemm2(mkdesc(h2_16, 1024, 1024, nullptr, 0, 0, w_g3i, g3bi,
                     nullptr, 0, gx32, 3072, 3072, 0), B_,
              mkdesc(h2_16, 1024, 1024, nullptr, 0, 0, w_g3h, g3bh,
                     nullptr, 0, gh32, 3072, 3072, 0), B_);
        k_gru<<<(B_ * 1024 + 255) / 256, 256, 0, stream>>>(gx32, gh32, h2_32, h32, h16,
                                                           H16 + (size_t)t * 1024, T_ * 1024);
    }

    // ---- decoder (time-parallel, chunked over M): theta heads ----------
    for (int c = 0; c < 4; ++c) {
        const f16* Za = Zp16 + (size_t)c * CM * 512;
        const f16* Ha = H16  + (size_t)c * CM * 1024;
        gemm1(mkdesc(Za, 512, 512, Ha, 1024, 1024, w_th1, thb1,
                     t1_16, 1024, nullptr, 0, 1024, 1), CM);
        gemm1(mkdesc(t1_16, 1024, 1024, nullptr, 0, 0, w_th2, thb2,
                     t2_16, 1024, nullptr, 0, 1024, 1), CM);
        gemm2(mkdesc(t2_16, 1024, 1024, nullptr, 0, 0, w_thm, thmb,
                     nullptr, 0, out_mu + (size_t)c * CM * 128, 128, 128, 0), CM,
              mkdesc(t2_16, 1024, 1024, nullptr, 0, 0, w_ths, thsb,
                     nullptr, 0, out_sig + (size_t)c * CM * 128, 128, 128, 2), CM);
    }
}
